// Sender_26396869001630
// MI455X (gfx1250) — matrix operand order repaired and checked by reading, not compiled, not run on hardware
//
#include <hip/hip_runtime.h>
#include <hip/hip_bf16.h>
#include <math.h>

// ---------------------------------------------------------------------------
// Sender GRU pipeline for MI455X (gfx1250, wave32, WMMA bf16->f32)
// ---------------------------------------------------------------------------

typedef __bf16 bf16;
typedef __attribute__((ext_vector_type(16))) __bf16 v16bf;
typedef __attribute__((ext_vector_type(8)))  float  v8f;

#define BATCH 8192
#define VIS   512
#define HID   256
#define EMB   64
#define VOC   4096
#define G3    (3 * HID)   // 768
#define TSTEPS 6
#define CHROWS 2048       // logits batch chunk

// ---------------------------------------------------------------------------
// float -> bf16 conversion (row-major, for A-side operands)
// ---------------------------------------------------------------------------
__global__ void f32_to_bf16_kernel(const float* __restrict__ src,
                                   bf16* __restrict__ dst, int n) {
  int i = blockIdx.x * blockDim.x + threadIdx.x;
  if (i < n) dst[i] = (bf16)src[i];
}

__global__ void zero_bf16_kernel(bf16* __restrict__ dst, int n) {
  int i = blockIdx.x * blockDim.x + threadIdx.x;
  if (i < n) dst[i] = (bf16)0.0f;
}

// ---------------------------------------------------------------------------
// Pack a K x N f32 matrix into WMMA B-fragment layout (bf16):
//   dst[((kt*N + col)*2 + half)*16 + e] = src[(kt*32 + half*16 + e)*N + col]
// so each lane's 16 K-consecutive values for one column are 32 contiguous
// bytes -> a B fragment becomes two global_load_b128 per lane.
// ---------------------------------------------------------------------------
__global__ void pack_b_kernel(const float* __restrict__ src,
                              bf16* __restrict__ dst, int K, int N) {
  int i = blockIdx.x * blockDim.x + threadIdx.x;   // over (K/32)*N*2 groups
  int total = (K >> 5) * N * 2;
  if (i >= total) return;
  int half = i & 1;
  int col  = (i >> 1) % N;
  int kt   = (i >> 1) / N;
  const float* s = src + (size_t)(kt * 32 + half * 16) * N + col;
  bf16* d = dst + (size_t)i * 16;
#pragma unroll
  for (int e = 0; e < 16; ++e) d[e] = (bf16)s[(size_t)e * N];
}

// ---------------------------------------------------------------------------
// Fragment loads (wave32). Contiguous k order (k = 16*half + s) for A and B, not the ISA table order.
// A 16x32 bf16 from row-major: q0 = K kb..kb+7, q1 = K kb+8..kb+15,
//   kb = +0 (lanes<16) / +16, row = row0 + (lane&15).
// ---------------------------------------------------------------------------
__device__ __forceinline__ v16bf ld_a_frag(const bf16* __restrict__ p) {
  union { uint4 q[2]; v16bf v; } u;
  u.q[0] = *reinterpret_cast<const uint4*>(p);
  u.q[1] = *reinterpret_cast<const uint4*>(p + 8);
  return u.v;
}

// B fragment from packed layout: 32 contiguous bytes per lane.
__device__ __forceinline__ v16bf ld_b_frag(const bf16* __restrict__ p) {
  union { uint4 q[2]; v16bf v; } u;
  u.q[0] = reinterpret_cast<const uint4*>(p)[0];
  u.q[1] = reinterpret_cast<const uint4*>(p)[1];
  return u.v;
}

#define WMMA_BF16(A_, B_, C_) \
  __builtin_amdgcn_wmma_f32_16x16x32_bf16(false, (A_), false, (B_), (short)0, (C_), false, false)

// ---------------------------------------------------------------------------
// GEMM: C = A[MxK] @ B[KxN] + bias, A row-major bf16, B pre-packed bf16.
// Wave tile 32x64: 2 A-frags x 4 B-frags -> 8 WMMA per 32-K step.
// Block 256 = 8 waves, arranged 4(M) x 2(N) -> 128x128 block tile.
// ---------------------------------------------------------------------------
template <bool WF, bool WB>
__global__ __launch_bounds__(256)
void gemm_bf16_kernel(const bf16* __restrict__ A, const bf16* __restrict__ Bp,
                      const float* __restrict__ bias,
                      float* __restrict__ Cf, bf16* __restrict__ Cb,
                      int M, int N, int K) {
  const int wave = threadIdx.x >> 5;
  const int lane = threadIdx.x & 31;
  const int row0 = blockIdx.y * 128 + (wave >> 1) * 32;
  const int col0 = blockIdx.x * 128 + (wave & 1) * 64;
  const int clane = lane & 15;
  const int half  = (lane >> 4) & 1;

  // per-lane streaming pointers
  const bf16* pA0 = A + (size_t)(row0 + clane) * K + half * 16;   // rows row0..+15
  const bf16* pA1 = pA0 + (size_t)16 * K;                        // rows row0+16..+31
  const bf16* pB  = Bp + (((size_t)col0 + clane) * 2 + half) * 16;
  const size_t bStep = (size_t)N * 32;

  v8f acc[2][4] = {};
  for (int kt = 0; kt < (K >> 5); ++kt) {
    v16bf a0 = ld_a_frag(pA0);
    v16bf a1 = ld_a_frag(pA1);
#pragma unroll
    for (int t = 0; t < 4; ++t) {
      v16bf b = ld_b_frag(pB + (size_t)t * 512);   // col tiles: +16 cols = +512 elems
      acc[0][t] = WMMA_BF16(a0, b, acc[0][t]);
      acc[1][t] = WMMA_BF16(a1, b, acc[1][t]);
    }
    pA0 += 32; pA1 += 32; pB += bStep;
  }

  // bias, hoisted
  float bs[4];
#pragma unroll
  for (int t = 0; t < 4; ++t) bs[t] = bias[col0 + 16 * t + clane];

  // C/D layout: VGPR j, lanes 0-15 -> row rbase+j, lanes 16-31 -> rbase+8+j.
#pragma unroll
  for (int r = 0; r < 2; ++r) {
    const int rbase = row0 + r * 16 + half * 8;
#pragma unroll
    for (int t = 0; t < 4; ++t) {
      const int ccol = col0 + 16 * t + clane;
      size_t idx = (size_t)rbase * N + ccol;
#pragma unroll
      for (int j = 0; j < 8; ++j) {
        const float v = acc[r][t][j] + bs[t];
        if (WF) Cf[idx] = v;
        if (WB) Cb[idx] = (bf16)v;
        idx += N;
      }
    }
  }
}

// ---------------------------------------------------------------------------
// GRU gates (elementwise).  hm is loop-invariant (the reference always feeds
// prev_hidden as the GRU hidden input).  Writes h f32 (d_out h_t region) and
// a bf16 copy for the logits GEMM.
// ---------------------------------------------------------------------------
__global__ void gru_gate_kernel(const float* __restrict__ xm,
                                const float* __restrict__ hm,
                                const float* __restrict__ hprev,
                                float* __restrict__ h_out,
                                bf16* __restrict__ h_bf, int n) {
  int i = blockIdx.x * blockDim.x + threadIdx.x;
  if (i >= n) return;
  int b = i / HID, c = i % HID;
  size_t r3 = (size_t)b * G3;
  float z  = 1.0f / (1.0f + __expf(-(xm[r3 + c] + hm[r3 + c])));
  float r  = 1.0f / (1.0f + __expf(-(xm[r3 + HID + c] + hm[r3 + HID + c])));
  float hh = tanhf(xm[r3 + 2 * HID + c] + r * hm[r3 + 2 * HID + c]);
  float h  = z * hprev[i] + (1.0f - z) * hh;
  h_out[i] = h;
  h_bf[i]  = (bf16)h;
}

// ---------------------------------------------------------------------------
// Fused log-softmax statistics: one block (256 thr) per batch row.
// Online merge of (max m, s = sum e^{l-m}, d = sum l*e^{l-m}, argmax):
//   sym = argmax, sel = -log(s), ent = (m + log s) - d/s.
// Accumulates entropy across steps; gathers emb[sym] as next bf16 input.
// ---------------------------------------------------------------------------
__global__ __launch_bounds__(256)
void softmax_stats_kernel(const float* __restrict__ logits,   // [CHROWS, VOC]
                          const bf16* __restrict__ emb_bf,    // [VOC, EMB]
                          bf16* __restrict__ x_bf,            // [BATCH, EMB]
                          float* __restrict__ seq, float* __restrict__ lg,
                          float* __restrict__ ent_acc,
                          float* __restrict__ ent_out, float* __restrict__ mlen,
                          int row_base, int t) {
  const int rb = blockIdx.x;
  const int b  = row_base + rb;
  const float* L = logits + (size_t)rb * VOC;

  float m = -INFINITY, s = 0.0f, d = 0.0f;
  int bi = 0;
  for (int i = threadIdx.x; i < VOC; i += blockDim.x) {
    float l = L[i];
    if (l > m) {
      float sc = __expf(m - l);
      s = s * sc + 1.0f; d = d * sc + l; m = l; bi = i;
    } else {
      float e = __expf(l - m);
      s += e; d += l * e;
    }
  }
#pragma unroll
  for (int off = 1; off < 32; off <<= 1) {
    float m2 = __shfl_xor(m, off, 32);
    float s2 = __shfl_xor(s, off, 32);
    float d2 = __shfl_xor(d, off, 32);
    int  bi2 = __shfl_xor(bi, off, 32);
    if (m2 > m) { float sc = __expf(m - m2); s = s * sc + s2; d = d * sc + d2; m = m2; bi = bi2; }
    else        { float sc = __expf(m2 - m); s += s2 * sc; d += d2 * sc; }
  }

  __shared__ float sm[8], ss[8], sd[8];
  __shared__ int   sbi[8];
  const int wave = threadIdx.x >> 5, lane = threadIdx.x & 31;
  if (lane == 0) { sm[wave] = m; ss[wave] = s; sd[wave] = d; sbi[wave] = bi; }
  __syncthreads();
  if (threadIdx.x == 0) {
    float M = sm[0], S = ss[0], D = sd[0];
    int BI = sbi[0];
    for (int w = 1; w < 8; ++w) {
      if (sm[w] > M) { float sc = __expf(M - sm[w]); S = S * sc + ss[w]; D = D * sc + sd[w]; M = sm[w]; BI = sbi[w]; }
      else           { float sc = __expf(sm[w] - M); S += ss[w] * sc; D += sd[w] * sc; }
    }
    float logS = __logf(S);
    float ent  = (M + logS) - D / S;
    seq[(size_t)b * TSTEPS + t] = (float)BI;
    lg[(size_t)b * TSTEPS + t]  = -logS;
    float tot = (t == 0 ? 0.0f : ent_acc[b]) + ent;
    if (t == TSTEPS - 1) { ent_out[b] = tot * (1.0f / TSTEPS); mlen[b] = (float)TSTEPS; }
    else                 { ent_acc[b] = tot; }
    sbi[0] = BI;
  }
  __syncthreads();
  const int BI = sbi[0];
  if (threadIdx.x < EMB)
    x_bf[(size_t)b * EMB + threadIdx.x] = emb_bf[(size_t)BI * EMB + threadIdx.x];
}

// ---------------------------------------------------------------------------
// Host-side orchestration
// ---------------------------------------------------------------------------
static inline size_t align256(size_t x) { return (x + 255) & ~(size_t)255; }

extern "C" void kernel_launch(void* const* d_in, const int* in_sizes, int n_in,
                              void* d_out, int out_size, void* d_ws, size_t ws_size,
                              hipStream_t stream) {
  (void)in_sizes; (void)n_in; (void)out_size; (void)ws_size;

  const float* vision = (const float*)d_in[0];   // [B, VIS]
  const float* Wv     = (const float*)d_in[1];   // [VIS, H]
  const float* bv     = (const float*)d_in[2];   // [H]
  const float* gk     = (const float*)d_in[3];   // [E, 3H]
  const float* grk    = (const float*)d_in[4];   // [H, 3H]
  const float* b_in   = (const float*)d_in[5];   // [3H]
  const float* b_rec  = (const float*)d_in[6];   // [3H]
  const float* Wo     = (const float*)d_in[7];   // [H, V]
  const float* bo     = (const float*)d_in[8];   // [V]
  const float* emb    = (const float*)d_in[9];   // [V, E]

  // outputs, concatenated flat
  float* seq     = (float*)d_out;                       // [B, T]
  float* lg      = seq + (size_t)BATCH * TSTEPS;        // [B, T]
  float* ent_out = lg + (size_t)BATCH * TSTEPS;         // [B]
  float* mlen    = ent_out + BATCH;                     // [B]
  float* ht      = mlen + BATCH;                        // [B, H]

  // workspace carve (byte cursor, 256B aligned)
  char* ws = (char*)d_ws;
  size_t off = 0;
  auto carve = [&](size_t bytes) { char* p = ws + off; off = align256(off + bytes); return p; };
  bf16*  vis_bf   = (bf16*)carve((size_t)BATCH * VIS * 2);   // A: vision
  bf16*  Wv_pk    = (bf16*)carve((size_t)VIS * HID * 2);     // B packed
  bf16*  gk_pk    = (bf16*)carve((size_t)EMB * G3 * 2);      // B packed
  bf16*  grk_pk   = (bf16*)carve((size_t)HID * G3 * 2);      // B packed
  bf16*  Wo_pk    = (bf16*)carve((size_t)HID * VOC * 2);     // B packed
  bf16*  emb_bf   = (bf16*)carve((size_t)VOC * EMB * 2);     // gather source
  float* hprev_f  = (float*)carve((size_t)BATCH * HID * 4);
  bf16*  hprev_bf = (bf16*)carve((size_t)BATCH * HID * 2);
  float* hm_f     = (float*)carve((size_t)BATCH * G3 * 4);
  float* xm_f     = (float*)carve((size_t)BATCH * G3 * 4);
  bf16*  x_bf     = (bf16*)carve((size_t)BATCH * EMB * 2);
  bf16*  h_bf     = (bf16*)carve((size_t)BATCH * HID * 2);
  float* logit_f  = (float*)carve((size_t)CHROWS * VOC * 4);
  float* ent_acc  = (float*)carve((size_t)BATCH * 4);

  // 1) one-time conversions: A-side row-major bf16, B-side WMMA-packed bf16
  auto conv = [&](const float* s, bf16* dst, int n) {
    f32_to_bf16_kernel<<<(n + 255) / 256, 256, 0, stream>>>(s, dst, n);
  };
  auto pack = [&](const float* s, bf16* dst, int K, int N) {
    int groups = (K >> 5) * N * 2;
    pack_b_kernel<<<(groups + 255) / 256, 256, 0, stream>>>(s, dst, K, N);
  };
  conv(vision, vis_bf, BATCH * VIS);
  conv(emb,    emb_bf, VOC * EMB);
  pack(Wv,  Wv_pk,  VIS, HID);
  pack(gk,  gk_pk,  EMB, G3);
  pack(grk, grk_pk, HID, G3);
  pack(Wo,  Wo_pk,  HID, VOC);
  // cell0 = zeros
  zero_bf16_kernel<<<(BATCH * EMB + 255) / 256, 256, 0, stream>>>(x_bf, BATCH * EMB);

  // 2) prev_hidden = vision @ Wv + bv  (f32 + bf16 copies)
  gemm_bf16_kernel<true, true><<<dim3(HID / 128, BATCH / 128), 256, 0, stream>>>(
      vis_bf, Wv_pk, bv, hprev_f, hprev_bf, BATCH, HID, VIS);

  // 3) hm = prev_hidden @ gru_rec_kernel + b_rec  (loop-invariant!)
  gemm_bf16_kernel<true, false><<<dim3(G3 / 128, BATCH / 128), 256, 0, stream>>>(
      hprev_bf, grk_pk, b_rec, hm_f, nullptr, BATCH, G3, HID);

  // 4) T decoding steps
  for (int t = 0; t < TSTEPS; ++t) {
    // xm = x @ gru_kernel + b_in   (K = 64)
    gemm_bf16_kernel<true, false><<<dim3(G3 / 128, BATCH / 128), 256, 0, stream>>>(
        x_bf, gk_pk, b_in, xm_f, nullptr, BATCH, G3, EMB);
    // GRU gates -> h (f32 into d_out h_t region; last step survives) + bf16 copy
    gru_gate_kernel<<<(BATCH * HID + 255) / 256, 256, 0, stream>>>(
        xm_f, hm_f, hprev_f, ht, h_bf, BATCH * HID);
    // logits = h @ Wo + bo, chunked over batch rows, then fused softmax stats
    for (int c = 0; c < BATCH / CHROWS; ++c) {
      gemm_bf16_kernel<true, false><<<dim3(VOC / 128, CHROWS / 128), 256, 0, stream>>>(
          h_bf + (size_t)c * CHROWS * HID, Wo_pk, bo, logit_f, nullptr,
          CHROWS, VOC, HID);
      softmax_stats_kernel<<<CHROWS, 256, 0, stream>>>(
          logit_f, emb_bf, x_bf, seq, lg, ent_acc, ent_out, mlen,
          c * CHROWS, t);
    }
  }
}
